// MultiHeadSelfAttention2D_69818988364146
// MI455X (gfx1250) — matrix operand order repaired and checked by reading, not compiled, not run on hardware
//
#include <hip/hip_runtime.h>
#include <stdint.h>

// ---------------- problem constants ----------------
#define H_   4
#define B_   2
#define C_   64
#define T_   2048
#define F_   64
#define HID_ 4
#define VD_  16
#define DQK  256          // HID*F
#define DV   1024         // VD*F
#define HB_  8            // H*B
#define EPS  1e-5f

// ---------------- vector types ----------------
typedef __attribute__((ext_vector_type(16))) __bf16 v16bf;
typedef __attribute__((ext_vector_type(8)))  __bf16 v8bf;
typedef __attribute__((ext_vector_type(8)))  float  v8f;
typedef __attribute__((ext_vector_type(4)))  unsigned int v4u;
typedef __attribute__((ext_vector_type(8)))  int v8i;
typedef __attribute__((ext_vector_type(4)))  int v4i;

// Splice two 16B half-fragments into one WMMA A-fragment (per lane,
// A holds k {0..15} (lanes<16) / {16..31} (lanes>=16), matching B).
__device__ __forceinline__ v16bf combine16(v8bf lo, v8bf hi) {
  v16bf r;
#pragma unroll
  for (int i = 0; i < 8; ++i) { r[i] = lo[i]; r[i + 8] = hi[i]; }
  return r;
}

// Block-wide sum of two values (256 threads).
__device__ __forceinline__ void bsum2(float a, float b, float* red,
                                      float* oa, float* ob) {
  int tid = threadIdx.x;
  red[tid] = a; red[256 + tid] = b;
  __syncthreads();
  for (int s = 128; s > 0; s >>= 1) {
    if (tid < s) { red[tid] += red[tid + s]; red[256 + tid] += red[256 + tid + s]; }
    __syncthreads();
  }
  *oa = red[0]; *ob = red[256];
  __syncthreads();
}

// =====================================================================
// Kernel 1: per-head 1x1-conv Q/K/V + PReLU + LayerNorm over (chan,freq).
// One block per (t,b); x slice (64x64) staged in LDS. Q gets the 1/sqrt(256)
// attention scale folded in. Outputs bf16 in (HB, T, D) row-major layout.
// =====================================================================
__global__ void __launch_bounds__(256) qkv_ln_kernel(
    const float* __restrict__ x,
    const float* __restrict__ Wq, const float* __restrict__ bq,
    const float* __restrict__ aq, const float* __restrict__ gq,
    const float* __restrict__ betq,
    const float* __restrict__ Wk, const float* __restrict__ bk,
    const float* __restrict__ ak, const float* __restrict__ gk,
    const float* __restrict__ betk,
    const float* __restrict__ Wv, const float* __restrict__ bv,
    const float* __restrict__ av, const float* __restrict__ gv,
    const float* __restrict__ betv,
    __bf16* __restrict__ Qo, __bf16* __restrict__ Ko, __bf16* __restrict__ Vo) {
  __shared__ float xs[C_][F_ + 2];
  __shared__ float red[512];
  int t = blockIdx.x, b = blockIdx.y, tid = threadIdx.x;

  for (int idx = tid; idx < C_ * F_; idx += 256) {
    int c = idx >> 6, f = idx & 63;
    xs[c][f] = x[(((size_t)b * C_ + c) * T_ + t) * F_ + f];
  }
  __syncthreads();

  int o = tid >> 6, f = tid & 63;           // Q/K: one (o,f) per thread per head
  for (int h = 0; h < H_; ++h) {
    // ---- Q and K (group size HID*F = 256 = blockDim) ----
    float yq = bq[h * HID_ + o], yk = bk[h * HID_ + o];
    const float* wq = Wq + (h * HID_ + o) * C_;
    const float* wk = Wk + (h * HID_ + o) * C_;
#pragma unroll 8
    for (int c = 0; c < C_; ++c) {
      float xv = xs[c][f];
      yq = fmaf(wq[c], xv, yq);
      yk = fmaf(wk[c], xv, yk);
    }
    yq = yq >= 0.f ? yq : aq[h] * yq;
    yk = yk >= 0.f ? yk : ak[h] * yk;

    float sa, sb;
    bsum2(yq, yq * yq, red, &sa, &sb);
    {
      float mu = sa * (1.f / 256.f), var = sb * (1.f / 256.f) - mu * mu;
      float rs = rsqrtf(var + EPS);
      float qn = (yq - mu) * rs * gq[(h * HID_ + o) * F_ + f] + betq[(h * HID_ + o) * F_ + f];
      Qo[(((size_t)(h * B_ + b)) * T_ + t) * DQK + tid] = (__bf16)(qn * 0.0625f);
    }
    bsum2(yk, yk * yk, red, &sa, &sb);
    {
      float mu = sa * (1.f / 256.f), var = sb * (1.f / 256.f) - mu * mu;
      float rs = rsqrtf(var + EPS);
      float kn = (yk - mu) * rs * gk[(h * HID_ + o) * F_ + f] + betk[(h * HID_ + o) * F_ + f];
      Ko[(((size_t)(h * B_ + b)) * T_ + t) * DQK + tid] = (__bf16)kn;
    }

    // ---- V (group size VD*F = 1024 -> 4 per thread) ----
    float yv[4], s = 0.f, ss = 0.f;
#pragma unroll
    for (int r = 0; r < 4; ++r) {
      int j = tid + r * 256;
      int vo = j >> 6, vf = j & 63;
      float y = bv[h * VD_ + vo];
      const float* wv = Wv + (h * VD_ + vo) * C_;
#pragma unroll 8
      for (int c = 0; c < C_; ++c) y = fmaf(wv[c], xs[c][vf], y);
      y = y >= 0.f ? y : av[h] * y;
      yv[r] = y; s += y; ss += y * y;
    }
    bsum2(s, ss, red, &sa, &sb);
    float mu = sa * (1.f / 1024.f), var = sb * (1.f / 1024.f) - mu * mu;
    float rs = rsqrtf(var + EPS);
#pragma unroll
    for (int r = 0; r < 4; ++r) {
      int j = tid + r * 256;
      int vo = j >> 6, vf = j & 63;
      float vn = (yv[r] - mu) * rs * gv[(h * VD_ + vo) * F_ + vf] + betv[(h * VD_ + vo) * F_ + vf];
      Vo[(((size_t)(h * B_ + b)) * T_ + t) * DV + j] = (__bf16)vn;
    }
  }
}

// =====================================================================
// Kernel 2: S = Q * K^T   (per hb; 128x128 tile per block, 8 waves).
// Q/K tiles (128 x 256 bf16) DMA'd to LDS via the Tensor Data Mover with
// pad_enable (4 dwords every 128 dwords) -> LDS row stride 264 halves for
// conflict-free fragment reads. 64 v_wmma_f32_16x16x32_bf16 per wave.
// =====================================================================
#define LDQ 264  // padded LDS row stride in halves (256 + 8)

__device__ __forceinline__ void tdm_load_tile_128x256(uint64_t ga, unsigned lds_off) {
  // D# group0: count=1, lds_addr, 57-bit global addr, type=2 ("image")
  v4u g0;
  g0[0] = 1u;
  g0[1] = lds_off;
  g0[2] = (unsigned)(ga & 0xFFFFFFFFu);
  g0[3] = ((unsigned)((ga >> 32) & 0x1FFFFFFu)) | 0x80000000u;
  // D# group1: data_size=2B, pad_enable, pad_interval=128 dwords (code 6),
  // pad_amount=4 dwords (code 3); tensor_dim0=256, tensor_dim1=128,
  // tile_dim0=256, tile_dim1=128, tensor_dim0_stride=256 elements.
  v8i g1;
  g1[0] = (1 << 16) | (1 << 20) | (6 << 22) | (3 << 25);
  g1[1] = (int)(256u << 16);   // tensor_dim0[15:0]
  g1[2] = (int)(128u << 16);   // tensor_dim1[15:0]
  g1[3] = (int)(256u << 16);   // tile_dim0
  g1[4] = 128;                 // tile_dim1
  g1[5] = 256;                 // tensor_dim0_stride lo
  g1[6] = 0;
  g1[7] = 0;
  v4i z4 = {0, 0, 0, 0};
  v8i z8 = {0, 0, 0, 0, 0, 0, 0, 0};
  // 6-arg form (amdgpu-toolchain / clang-23): groups 2,3 + extra group, cpol
  __builtin_amdgcn_tensor_load_to_lds(g0, g1, z4, z4, z8, 0);
}

__global__ void __launch_bounds__(256) qk_gemm_kernel(
    const __bf16* __restrict__ Q, const __bf16* __restrict__ K,
    float* __restrict__ S) {
  extern __shared__ __bf16 smem[];
  __bf16* sQ = smem;                 // 128 x LDQ halves (offset 0)
  __bf16* sK = smem + 128 * LDQ;     // 128 x LDQ halves

  int kt = blockIdx.x, qt = blockIdx.y, hb = blockIdx.z;
  int qbase = qt * 128, kbase = kt * 128;
  int tid = threadIdx.x, wave = tid >> 5, lane = tid & 31;
  int l15 = lane & 15, hh = lane >> 4;

  if (wave == 0) {
    tdm_load_tile_128x256((uint64_t)(uintptr_t)(Q + ((size_t)hb * T_ + qbase) * DQK), 0u);
    tdm_load_tile_128x256((uint64_t)(uintptr_t)(K + ((size_t)hb * T_ + kbase) * DQK),
                          (unsigned)(128 * LDQ * 2));
    __builtin_amdgcn_s_wait_tensorcnt(0);
  }
  __syncthreads();

  // A-fragments for this wave's 16 Q rows (reused over all 8 column tiles)
  int wrow = (wave << 4) + l15;
  v16bf afr[8];
#pragma unroll
  for (int d0 = 0; d0 < 8; ++d0) {
    const __bf16* ap = sQ + wrow * LDQ + d0 * 32 + hh * 16;
    afr[d0] = combine16(*(const v8bf*)ap, *(const v8bf*)(ap + 8));
  }

#pragma unroll
  for (int j = 0; j < 8; ++j) {
    v8f acc = {0.f, 0.f, 0.f, 0.f, 0.f, 0.f, 0.f, 0.f};
#pragma unroll
    for (int d0 = 0; d0 < 8; ++d0) {
      const __bf16* bp = sK + (16 * j + l15) * LDQ + d0 * 32 + hh * 16;
      v16bf bfr = *(const v16bf*)bp;
      acc = __builtin_amdgcn_wmma_f32_16x16x32_bf16(
          false, afr[d0], false, bfr, (short)0, acc, false, false);
    }
    float* sp = S + ((size_t)hb * T_ + qbase + (wave << 4) + (hh << 3)) * T_ +
                kbase + 16 * j + l15;
#pragma unroll
    for (int v = 0; v < 8; ++v) sp[(size_t)v * T_] = acc[v];
  }
}

// =====================================================================
// Kernel 3: row softmax of S (8 x 2048 rows of 2048 f32), output bf16 P.
// S/P (201 MB combined) stay resident in the 192MB L2 between kernels.
// =====================================================================
__global__ void __launch_bounds__(256) softmax_kernel(
    const float* __restrict__ S, __bf16* __restrict__ P) {
  __shared__ float red[256];
  int row = blockIdx.x, hb = blockIdx.y, tid = threadIdx.x;
  const float* sr = S + ((size_t)hb * T_ + row) * T_;
  float v[8], mx = -3.4e38f;
#pragma unroll
  for (int i = 0; i < 8; ++i) { v[i] = sr[tid + i * 256]; mx = fmaxf(mx, v[i]); }
  red[tid] = mx; __syncthreads();
  for (int s = 128; s > 0; s >>= 1) {
    if (tid < s) red[tid] = fmaxf(red[tid], red[tid + s]);
    __syncthreads();
  }
  mx = red[0]; __syncthreads();
  float sum = 0.f;
#pragma unroll
  for (int i = 0; i < 8; ++i) { v[i] = __expf(v[i] - mx); sum += v[i]; }
  red[tid] = sum; __syncthreads();
  for (int s = 128; s > 0; s >>= 1) {
    if (tid < s) red[tid] += red[tid + s];
    __syncthreads();
  }
  float inv = 1.f / red[0];
  __bf16* pr = P + ((size_t)hb * T_ + row) * T_;
#pragma unroll
  for (int i = 0; i < 8; ++i) pr[tid + i * 256] = (__bf16)(v[i] * inv);
}

// =====================================================================
// Kernel 4: O = P * V  (per hb; block tile 128 q-rows x 128 v-cols,
// k-loop over 2048 in chunks of 64). P chunk staged with async-to-LDS
// (global_load_async_to_lds_b128 + s_wait_asynccnt); V chunk transposed
// into LDS for B-fragments. Result written into (B, C, T, F) f32 layout.
// =====================================================================
#define LDP 72  // padded LDS row stride in halves (64 + 8)

__global__ void __launch_bounds__(256) pv_gemm_kernel(
    const __bf16* __restrict__ P, const __bf16* __restrict__ V,
    float* __restrict__ O) {
  __shared__ __bf16 sP[128 * LDP];    // P tile: 128 rows x 64 k (padded)
  __shared__ __bf16 sVt[128 * LDP];   // V^T tile: 128 d-rows x 64 k (padded)
  int dc = blockIdx.x, qt = blockIdx.y, hb = blockIdx.z;
  int dbase = dc * 128, qbase = qt * 128;
  int tid = threadIdx.x, wave = tid >> 5, lane = tid & 31;
  int l15 = lane & 15, hh = lane >> 4;

  const __bf16* Pg = P + ((size_t)hb * T_ + qbase) * T_;
  const __bf16* Vg = V + (size_t)hb * T_ * DV + dbase;

  v8f acc[8];
#pragma unroll
  for (int j = 0; j < 8; ++j) acc[j] = (v8f){0.f, 0.f, 0.f, 0.f, 0.f, 0.f, 0.f, 0.f};

  for (int k0 = 0; k0 < T_; k0 += 64) {
    __syncthreads();  // prior iteration's LDS readers done
    // ---- stage P[qbase.., k0..k0+64) via async-to-LDS (16B per lane) ----
#pragma unroll
    for (int r = 0; r < 4; ++r) {
      int idx = tid + r * 256;           // 1024 x 16B = 16KB raw rows
      int prow = idx >> 3, pch = idx & 7;
      uint64_t ga = (uint64_t)(uintptr_t)(Pg + (size_t)prow * T_ + k0 + pch * 8);
      unsigned lds = (unsigned)(uintptr_t)(&sP[prow * LDP + pch * 8]);
      asm volatile("global_load_async_to_lds_b128 %0, %1, off"
                   :: "v"(lds), "v"(ga) : "memory");
    }
    // ---- stage V[k0..k0+64, dbase..dbase+128) transposed ----
#pragma unroll
    for (int r = 0; r < 4; ++r) {
      int idx = tid + r * 256;
      int kk = idx >> 4, d8 = idx & 15;
      v8bf vv = *(const v8bf*)(Vg + (size_t)(k0 + kk) * DV + d8 * 8);
#pragma unroll
      for (int e = 0; e < 8; ++e) sVt[(d8 * 8 + e) * LDP + kk] = vv[e];
    }
    asm volatile("s_wait_asynccnt 0" ::: "memory");
    __syncthreads();

#pragma unroll
    for (int kk0 = 0; kk0 < 64; kk0 += 32) {
      const __bf16* ap = sP + ((wave << 4) + l15) * LDP + kk0 + hh * 16;
      v16bf afr = combine16(*(const v8bf*)ap, *(const v8bf*)(ap + 8));
#pragma unroll
      for (int j = 0; j < 8; ++j) {
        const __bf16* bp = sVt + (16 * j + l15) * LDP + kk0 + hh * 16;
        v16bf bfr = *(const v16bf*)bp;
        acc[j] = __builtin_amdgcn_wmma_f32_16x16x32_bf16(
            false, afr, false, bfr, (short)0, acc[j], false, false);
      }
    }
  }

  // epilogue: scatter to (B, H*VD, T, F); hb = h*B + b
  int b = hb & 1, h = hb >> 1;
#pragma unroll
  for (int j = 0; j < 8; ++j) {
    int d = dbase + 16 * j + l15;
    int oc = h * VD_ + (d >> 6), f = d & 63;
    int mbase = qbase + (wave << 4) + (hh << 3);
#pragma unroll
    for (int v = 0; v < 8; ++v)
      O[(((size_t)b * C_ + oc) * T_ + mbase + v) * F_ + f] = acc[j][v];
  }
}

// =====================================================================
// Kernel 5: y = LN(PReLU(Wp*O + bp)) * gp + betp + x   (per (b,t) slice)
// =====================================================================
__global__ void __launch_bounds__(256) proj_ln_kernel(
    const float* __restrict__ O, const float* __restrict__ Wp,
    const float* __restrict__ bp, const float* __restrict__ ap,
    const float* __restrict__ gp, const float* __restrict__ betp,
    const float* __restrict__ x, float* __restrict__ out) {
  __shared__ float Os[C_][F_ + 2];
  __shared__ float red[512];
  int t = blockIdx.x, b = blockIdx.y, tid = threadIdx.x;

  for (int idx = tid; idx < C_ * F_; idx += 256) {
    int c = idx >> 6, f = idx & 63;
    Os[c][f] = O[(((size_t)b * C_ + c) * T_ + t) * F_ + f];
  }
  __syncthreads();

  float a = ap[0];
  float y[16], s = 0.f, ss = 0.f;
#pragma unroll
  for (int r = 0; r < 16; ++r) {
    int j = tid + r * 256;
    int o = j >> 6, f = j & 63;
    float acc = bp[o];
    const float* w = Wp + o * C_;
#pragma unroll 8
    for (int c = 0; c < C_; ++c) acc = fmaf(w[c], Os[c][f], acc);
    acc = acc >= 0.f ? acc : a * acc;
    y[r] = acc; s += acc; ss += acc * acc;
  }
  float sa, sb;
  bsum2(s, ss, red, &sa, &sb);
  float mu = sa * (1.f / 4096.f), var = sb * (1.f / 4096.f) - mu * mu;
  float rs = rsqrtf(var + EPS);
#pragma unroll
  for (int r = 0; r < 16; ++r) {
    int j = tid + r * 256;
    int o = j >> 6, f = j & 63;
    size_t oi = (((size_t)b * C_ + o) * T_ + t) * F_ + f;
    out[oi] = (y[r] - mu) * rs * gp[o * F_ + f] + betp[o * F_ + f] + x[oi];
  }
}

// =====================================================================
// Host-side launcher
// =====================================================================
extern "C" void kernel_launch(void* const* d_in, const int* in_sizes, int n_in,
                              void* d_out, int out_size, void* d_ws, size_t ws_size,
                              hipStream_t stream) {
  (void)in_sizes; (void)n_in; (void)out_size; (void)ws_size;
  const float* x    = (const float*)d_in[0];
  const float* Wq   = (const float*)d_in[1];
  const float* bq   = (const float*)d_in[2];
  const float* aq   = (const float*)d_in[3];
  const float* gq   = (const float*)d_in[4];
  const float* betq = (const float*)d_in[5];
  const float* Wk   = (const float*)d_in[6];
  const float* bk   = (const float*)d_in[7];
  const float* ak   = (const float*)d_in[8];
  const float* gk   = (const float*)d_in[9];
  const float* betk = (const float*)d_in[10];
  const float* Wv   = (const float*)d_in[11];
  const float* bv   = (const float*)d_in[12];
  const float* av   = (const float*)d_in[13];
  const float* gv   = (const float*)d_in[14];
  const float* betv = (const float*)d_in[15];
  const float* Wp   = (const float*)d_in[16];
  const float* bp   = (const float*)d_in[17];
  const float* ap   = (const float*)d_in[18];
  const float* gp   = (const float*)d_in[19];
  const float* betp = (const float*)d_in[20];

  // Workspace layout (bytes): Q 8M | K 8M | V 32M | S 128M | P 64M | O 64M
  char* ws = (char*)d_ws;
  __bf16* Q = (__bf16*)(ws + 0);
  __bf16* K = (__bf16*)(ws + 8388608);
  __bf16* V = (__bf16*)(ws + 16777216);
  float*  S = (float*) (ws + 50331648);
  __bf16* P = (__bf16*)(ws + 184549376);
  float*  O = (float*) (ws + 251658240);

  dim3 blk(256);
  qkv_ln_kernel<<<dim3(T_, B_), blk, 0, stream>>>(
      x, Wq, bq, aq, gq, betq, Wk, bk, ak, gk, betk, Wv, bv, av, gv, betv,
      Q, K, V);
  qk_gemm_kernel<<<dim3(T_ / 128, T_ / 128, HB_), blk, 2 * 128 * LDQ * 2, stream>>>(Q, K, S);
  softmax_kernel<<<dim3(T_, HB_), blk, 0, stream>>>(S, P);
  pv_gemm_kernel<<<dim3(DV / 128, T_ / 128, HB_), blk, 0, stream>>>(P, V, O);
  proj_ln_kernel<<<dim3(T_, B_), blk, 0, stream>>>(O, Wp, bp, ap, gp, betp, x,
                                                   (float*)d_out);
}
